// NMF_Nodes_36919538876753
// MI455X (gfx1250) — matrix operand order repaired and checked by reading, not compiled, not run on hardware
//
#include <hip/hip_runtime.h>
#include <hip/hip_bf16.h>

// ---------- types ----------
typedef __attribute__((ext_vector_type(8)))  float  v8f;
typedef __attribute__((ext_vector_type(8)))  __bf16 v8bf;
typedef __attribute__((ext_vector_type(16))) __bf16 v16bf;

#define N_NODES 4096
#define DIM     512
#define TOPICS  10
#define NT      6

// =====================================================================
// CDNA5 async global->LDS copy (16B per lane), tracked by ASYNCcnt.
// VDST = per-lane LDS byte address, VADDR = 32-bit byte offset, SADDR = base.
// =====================================================================
__device__ __forceinline__ void async_ld16(unsigned ldsAddr, unsigned vOff,
                                           unsigned long long base) {
    asm volatile("global_load_async_to_lds_b128 %0, %1, %2"
                 :: "v"(ldsAddr), "v"(vOff), "s"(base) : "memory");
}
__device__ __forceinline__ void wait_async0() {
    asm volatile("s_wait_asynccnt 0x0" ::: "memory");
}
__device__ __forceinline__ unsigned lds_off(const void* p) {
    // generic LDS pointer: low 32 bits are the LDS byte address
    return (unsigned)(unsigned long long)p;
}

// =====================================================================
// Kernel 0: zero the column-sum accumulator (graph-safe re-init)
// =====================================================================
__global__ void zero_kernel(float* __restrict__ colsum) {
    int i = blockIdx.x * 256 + threadIdx.x;
    if (i < N_NODES) colsum[i] = 0.0f;
}

// =====================================================================
// Kernel 1: per-row prep.  One block per row n:
//   - W[t,i] = sum_d x[n,d]*H[d,t,i]; code[n] = packed argmax_t (4b x 6)
//   - inv_norm = 1/max(||x_n||,1e-8)
//   - xnH/xnL = bf16 hi/lo split of xn (for bf16x3 cosine GEMM)
//   - xT[d][n] = bf16(x[n,d])  (transposed, un-normalized, GEMM-2 B)
// =====================================================================
__global__ __launch_bounds__(256) void prep_kernel(
        const float* __restrict__ x, const float* __restrict__ H,
        __bf16* __restrict__ xnH, __bf16* __restrict__ xnL,
        __bf16* __restrict__ xT, unsigned* __restrict__ codes) {
    __shared__ float xs[DIM];
    __shared__ float part[4 * 64];
    __shared__ float Wv[64];
    __shared__ float s_inv;

    const int n = blockIdx.x;
    const int t = threadIdx.x;

    xs[t]       = x[(size_t)n * DIM + t];
    xs[t + 256] = x[(size_t)n * DIM + t + 256];
    __syncthreads();

    const int o = t & 63;   // 0..59 = W slots, 60 = norm^2
    const int q = t >> 6;
    if (o < 61) {
        float s = 0.0f;
        const int d0 = q * 128;
        if (o < 60) {
            #pragma unroll 4
            for (int d = d0; d < d0 + 128; ++d) s += xs[d] * H[(size_t)d * 60 + o];
        } else {
            #pragma unroll 4
            for (int d = d0; d < d0 + 128; ++d) s += xs[d] * xs[d];
        }
        part[q * 64 + o] = s;
    }
    __syncthreads();

    if (t < 61) {
        float s = part[t] + part[64 + t] + part[128 + t] + part[192 + t];
        if (t < 60) Wv[t] = s;
        else        s_inv = 1.0f / fmaxf(sqrtf(s), 1e-8f);
    }
    __syncthreads();

    if (t == 0) {
        unsigned code = 0;
        #pragma unroll
        for (int i = 0; i < NT; ++i) {
            float best = Wv[i];
            int bi = 0;
            #pragma unroll
            for (int tt = 1; tt < TOPICS; ++tt) {
                float w = Wv[tt * NT + i];
                if (w > best) { best = w; bi = tt; }   // first-max tie-break
            }
            code |= ((unsigned)bi) << (4 * i);
        }
        codes[n] = code;
    }

    const float inv = s_inv;
    #pragma unroll
    for (int half = 0; half < 2; ++half) {
        const int d = t + half * 256;
        float v = xs[d] * inv;
        __bf16 h = (__bf16)v;
        __bf16 l = (__bf16)(v - (float)h);
        xnH[(size_t)n * DIM + d] = h;
        xnL[(size_t)n * DIM + d] = l;
        xT[(size_t)d * N_NODES + n] = (__bf16)xs[d];
    }
}

// =====================================================================
// Kernel 2: C = F * (xn @ xn^T) via bf16x3 WMMA (16x16x32), async-LDS
// double buffering. 128x128 macro-tile per block (grid 32x32), 8 waves,
// each wave = 2x4 WMMA tiles (32x64 region). Per 32-wide K-slab:
//   - prefetch next slab with GLOBAL_LOAD_ASYNC_TO_LDS_B128 (ASYNCcnt)
//   - 24 bf16 WMMAs: hi*hi + hi*lo + lo*hi (~fp32 accuracy, f32 accum)
// Epilogue: F-scale via packed-code nibble match, bf16 C store, colsum
// atomics.
// =====================================================================
#define LDB 40   // bf16 row stride: 80B rows -> conflict-free b128 frags

__global__ __launch_bounds__(256) void simgemm_kernel(
        const __bf16* __restrict__ xnH, const __bf16* __restrict__ xnL,
        const unsigned* __restrict__ codes, __bf16* __restrict__ Cbf,
        float* __restrict__ colsum) {
    __shared__ __bf16 sAH[2][128 * LDB];
    __shared__ __bf16 sAL[2][128 * LDB];
    __shared__ __bf16 sBH[2][128 * LDB];
    __shared__ __bf16 sBL[2][128 * LDB];
    __shared__ unsigned cA[128];
    __shared__ unsigned cB[128];

    const int aRow0 = blockIdx.y * 128;
    const int bCol0 = blockIdx.x * 128;
    const int tid    = threadIdx.x;
    const int lane   = tid & 31;
    const int w      = tid >> 5;
    const int laneHi = lane >> 4;
    const int laneLo = lane & 15;
    const int trBase = (w >> 1) * 2;   // tile-row start: 0,2,4,6
    const int tcBase = (w & 1) * 4;    // tile-col start: 0,4

    if (tid < 128) cA[tid]       = codes[aRow0 + tid];
    else           cB[tid - 128] = codes[bCol0 + tid - 128];

    const unsigned long long baseAH = (unsigned long long)(xnH + (size_t)aRow0 * DIM);
    const unsigned long long baseAL = (unsigned long long)(xnL + (size_t)aRow0 * DIM);
    const unsigned long long baseBH = (unsigned long long)(xnH + (size_t)bCol0 * DIM);
    const unsigned long long baseBL = (unsigned long long)(xnL + (size_t)bCol0 * DIM);

    // per-thread chunk coords: 512 chunks of 16B per plane per slab
    const int r0 = tid >> 2;                 // chunk tid
    const int c0 = (tid & 3) * 8;
    const int r1 = (tid + 256) >> 2;         // chunk tid+256
    const int c1 = ((tid + 256) & 3) * 8;

    v8f acc[2][4];
    #pragma unroll
    for (int i = 0; i < 2; ++i)
        #pragma unroll
        for (int j = 0; j < 4; ++j) acc[i][j] = (v8f)0.0f;

    // ---- issue one slab's async copies into buffer `buf` ----
    #define ISSUE_SLAB(k0, buf)                                                   \
        do {                                                                      \
            unsigned vo0 = (unsigned)(((r0) * DIM + (k0) + (c0)) * 2);            \
            unsigned vo1 = (unsigned)(((r1) * DIM + (k0) + (c1)) * 2);            \
            unsigned d0 = (unsigned)(((r0) * LDB + (c0)) * 2);                    \
            unsigned d1 = (unsigned)(((r1) * LDB + (c1)) * 2);                    \
            async_ld16(lds_off(&sAH[buf][0]) + d0, vo0, baseAH);                  \
            async_ld16(lds_off(&sAH[buf][0]) + d1, vo1, baseAH);                  \
            async_ld16(lds_off(&sAL[buf][0]) + d0, vo0, baseAL);                  \
            async_ld16(lds_off(&sAL[buf][0]) + d1, vo1, baseAL);                  \
            async_ld16(lds_off(&sBH[buf][0]) + d0, vo0, baseBH);                  \
            async_ld16(lds_off(&sBH[buf][0]) + d1, vo1, baseBH);                  \
            async_ld16(lds_off(&sBL[buf][0]) + d0, vo0, baseBL);                  \
            async_ld16(lds_off(&sBL[buf][0]) + d1, vo1, baseBL);                  \
        } while (0)

    ISSUE_SLAB(0, 0);
    wait_async0();
    __syncthreads();

    for (int s = 0; s < DIM / 32; ++s) {
        const int buf = s & 1;
        if (s + 1 < DIM / 32) ISSUE_SLAB((s + 1) * 32, buf ^ 1);

        // ---- build fragments from current buffer ----
        v16bf aH[2], aL[2], bH[4], bL[4];
        #pragma unroll
        for (int tr = 0; tr < 2; ++tr) {
            const int row = (trBase + tr) * 16 + laneLo;
            const __bf16* pH = &sAH[buf][row * LDB + laneHi * 8];
            const __bf16* pL = &sAL[buf][row * LDB + laneHi * 8];
            v8bf h0 = *(const v8bf*)pH;         // K slots 0..7 of lane-half
            v8bf h1 = *(const v8bf*)(pH + 16);  // K slots 16..23 of lane-half
            v8bf l0 = *(const v8bf*)pL;
            v8bf l1 = *(const v8bf*)(pL + 16);
            aH[tr] = __builtin_shufflevector(h0, h1, 0,1,2,3,4,5,6,7,8,9,10,11,12,13,14,15);
            aL[tr] = __builtin_shufflevector(l0, l1, 0,1,2,3,4,5,6,7,8,9,10,11,12,13,14,15);
        }
        #pragma unroll
        for (int tc = 0; tc < 4; ++tc) {
            const int row = (tcBase + tc) * 16 + laneLo;
            const __bf16* pH = &sBH[buf][row * LDB + laneHi * 8];
            const __bf16* pL = &sBL[buf][row * LDB + laneHi * 8];
            v8bf h0 = *(const v8bf*)pH;         // K slots 0..7 / 16..23 of lane-half
            v8bf h1 = *(const v8bf*)(pH + 16);
            v8bf l0 = *(const v8bf*)pL;
            v8bf l1 = *(const v8bf*)(pL + 16);
            bH[tc] = __builtin_shufflevector(h0, h1, 0,1,2,3,4,5,6,7,8,9,10,11,12,13,14,15);
            bL[tc] = __builtin_shufflevector(l0, l1, 0,1,2,3,4,5,6,7,8,9,10,11,12,13,14,15);
        }

        // ---- bf16x3: hi*hi + hi*lo + lo*hi, fp32 accumulate ----
        #pragma unroll
        for (int tr = 0; tr < 2; ++tr) {
            #pragma unroll
            for (int tc = 0; tc < 4; ++tc) {
                acc[tr][tc] = __builtin_amdgcn_wmma_f32_16x16x32_bf16(
                    false, aH[tr], false, bH[tc], (short)0, acc[tr][tc], false, false);
                acc[tr][tc] = __builtin_amdgcn_wmma_f32_16x16x32_bf16(
                    false, aH[tr], false, bL[tc], (short)0, acc[tr][tc], false, false);
                acc[tr][tc] = __builtin_amdgcn_wmma_f32_16x16x32_bf16(
                    false, aL[tr], false, bH[tc], (short)0, acc[tr][tc], false, false);
            }
        }

        wait_async0();       // next slab landed (own wave's copies)
        __syncthreads();     // ...and everyone else's
    }
    #undef ISSUE_SLAB

    // ---- Epilogue: F-scale, bf16 store, column-sum atomics ----
    const float inv6 = 1.0f / 6.0f;
    #pragma unroll
    for (int tc = 0; tc < 4; ++tc) {
        const int nLoc = (tcBase + tc) * 16 + laneLo;
        const unsigned cb = cB[nLoc];
        float colp = 0.0f;
        #pragma unroll
        for (int tr = 0; tr < 2; ++tr) {
            v8f a = acc[tr][tc];
            #pragma unroll
            for (int j = 0; j < 8; ++j) {
                const int mLoc = (trBase + tr) * 16 + j + laneHi * 8;
                unsigned xr = cA[mLoc] ^ cb;
                xr |= xr >> 1; xr |= xr >> 2;           // nibble-OR into bit0
                float F = (float)__popc(~xr & 0x111111u) * inv6;
                float cval = a[j] * F;
                Cbf[(size_t)(aRow0 + mLoc) * N_NODES + (bCol0 + nLoc)] = (__bf16)cval;
                colp += cval;
            }
        }
        atomicAdd(&colsum[bCol0 + nLoc], colp);
    }
}

// =====================================================================
// Kernel 3: out_first[b,d] = x[b,d] * (colsum[b]/N)
// =====================================================================
__global__ void outfirst_kernel(const float* __restrict__ x,
                                const float* __restrict__ colsum,
                                float* __restrict__ out) {
    int idx = blockIdx.x * 256 + threadIdx.x;   // over N*D
    int b = idx >> 9;
    int d = idx & 511;
    out[(size_t)b * 1024 + d] = x[idx] * colsum[b] * (1.0f / N_NODES);
}

// =====================================================================
// Kernel 4: out_second = (C @ x)/N via bf16 WMMA 16x16x32 (C symmetric).
// A = Cbf row-major, B from xT (x transposed bf16): contiguous 16B frags.
// Grid (4, 32), 128x128 tile per block.
// =====================================================================
__global__ __launch_bounds__(256) void outsecond_kernel(
        const __bf16* __restrict__ Cbf, const __bf16* __restrict__ xT,
        float* __restrict__ out) {
    const int mRow0 = blockIdx.y * 128;
    const int nCol0 = blockIdx.x * 128;
    const int tid    = threadIdx.x;
    const int lane   = tid & 31;
    const int w      = tid >> 5;
    const int laneHi = lane >> 4;
    const int laneLo = lane & 15;
    const int trBase = (w >> 1) * 2;
    const int tcBase = (w & 1) * 4;

    v8f acc[2][4];
    #pragma unroll
    for (int i = 0; i < 2; ++i)
        #pragma unroll
        for (int j = 0; j < 4; ++j) acc[i][j] = (v8f)0.0f;

    for (int k0 = 0; k0 < N_NODES; k0 += 32) {
        v16bf afr[2];
        #pragma unroll
        for (int tr = 0; tr < 2; ++tr) {
            const int m = mRow0 + (trBase + tr) * 16 + laneLo;
            const __bf16* p = &Cbf[(size_t)m * N_NODES + k0 + laneHi * 8];
            v8bf lo = *(const v8bf*)p;
            v8bf hi = *(const v8bf*)(p + 16);
            afr[tr] = __builtin_shufflevector(lo, hi, 0,1,2,3,4,5,6,7,8,9,10,11,12,13,14,15);
        }
        v16bf bfr[4];
        #pragma unroll
        for (int tc = 0; tc < 4; ++tc) {
            const int nn = nCol0 + (tcBase + tc) * 16 + laneLo;
            const __bf16* p = &xT[(size_t)nn * N_NODES + k0 + laneHi * 8];
            v8bf lo = *(const v8bf*)p;
            v8bf hi = *(const v8bf*)(p + 16);
            bfr[tc] = __builtin_shufflevector(lo, hi, 0,1,2,3,4,5,6,7,8,9,10,11,12,13,14,15);
        }
        #pragma unroll
        for (int tr = 0; tr < 2; ++tr)
            #pragma unroll
            for (int tc = 0; tc < 4; ++tc)
                acc[tr][tc] = __builtin_amdgcn_wmma_f32_16x16x32_bf16(
                    false, afr[tr], false, bfr[tc], (short)0, acc[tr][tc], false, false);
    }

    #pragma unroll
    for (int tr = 0; tr < 2; ++tr) {
        #pragma unroll
        for (int tc = 0; tc < 4; ++tc) {
            v8f a = acc[tr][tc];
            const int n = nCol0 + (tcBase + tc) * 16 + laneLo;
            #pragma unroll
            for (int j = 0; j < 8; ++j) {
                const int m = mRow0 + (trBase + tr) * 16 + j + laneHi * 8;
                out[(size_t)m * 1024 + 512 + n] = a[j] * (1.0f / N_NODES);
            }
        }
    }
}

// =====================================================================
// Host-side launcher
// =====================================================================
extern "C" void kernel_launch(void* const* d_in, const int* in_sizes, int n_in,
                              void* d_out, int out_size, void* d_ws, size_t ws_size,
                              hipStream_t stream) {
    const float* x = (const float*)d_in[0];   // (4096, 512) f32
    const float* H = (const float*)d_in[1];   // (512, 10, 6) f32
    float* out = (float*)d_out;               // (4096, 1024) f32
    (void)in_sizes; (void)n_in; (void)out_size; (void)ws_size;

    char* ws = (char*)d_ws;
    __bf16*   xnH    = (__bf16*)ws;                                      // 4 MB
    __bf16*   xnL    = (__bf16*)(ws + (size_t)4  * 1024 * 1024);         // 4 MB
    __bf16*   xT     = (__bf16*)(ws + (size_t)8  * 1024 * 1024);         // 4 MB
    unsigned* codes  = (unsigned*)(ws + (size_t)12 * 1024 * 1024);       // 16 KB
    float*    colsum = (float*)(ws + (size_t)12 * 1024 * 1024 + 16384);  // 16 KB
    __bf16*   Cbf    = (__bf16*)(ws + (size_t)12 * 1024 * 1024 + 32768); // 32 MB

    zero_kernel<<<16, 256, 0, stream>>>(colsum);
    prep_kernel<<<N_NODES, 256, 0, stream>>>(x, H, xnH, xnL, xT, codes);
    simgemm_kernel<<<dim3(32, 32), 256, 0, stream>>>(xnH, xnL, codes, Cbf, colsum);
    outfirst_kernel<<<(N_NODES * DIM) / 256, 256, 0, stream>>>(x, colsum, out);
    outsecond_kernel<<<dim3(4, 32), 256, 0, stream>>>(Cbf, xT, out);
}
